// L2GTraversal_74088185856061
// MI455X (gfx1250) — matrix operand order repaired and checked by reading, not compiled, not run on hardware
//
#include <hip/hip_runtime.h>
#include <hip/hip_bf16.h>

// ---------------------------------------------------------------------------
// CDNA5 (gfx1250) wave32 WMMA types
// ---------------------------------------------------------------------------
typedef __attribute__((ext_vector_type(16))) __bf16 v16bf;
typedef __attribute__((ext_vector_type(8)))  float  v8f;

union Frag16 {
    v16bf v;
    unsigned short us[16];
    uint4 q[2];
};

__device__ __forceinline__ unsigned short f2bf(float f) {
    // round-to-nearest-even f32 -> bf16 (scalar, used only in tiny prep kernel)
    unsigned int u = __float_as_uint(f);
    unsigned int r = u + 0x7FFFu + ((u >> 16) & 1u);
    return (unsigned short)(r >> 16);
}

// Packed f32x2 -> bf16x2 via the CDNA5 HW instruction (no clang builtin for it;
// the __builtin_amdgcn_cvt_pk_bf16_f32 probe in round 2 fell through to the
// scalar path, so go through inline asm).  lo -> [15:0], hi -> [31:16].
__device__ __forceinline__ unsigned int cvt_pk_bf16(float lo, float hi) {
    unsigned int r;
    asm("v_cvt_pk_bf16_f32 %0, %1, %2" : "=v"(r) : "v"(lo), "v"(hi));
    return r;
}

// A fragment (16x32 bf16, MxK): lane L<16 holds row M=L, K={k0..k0+7, k0+16..k0+23};
// lane L>=16 holds row M=L-16, K={k0+8..k0+15, k0+24..k0+31}.   (ISA 7.12.2)
__device__ __forceinline__ v16bf load_frag_a(const unsigned short* rowp, int half) {
    Frag16 f;
    f.q[0] = *(const uint4*)(rowp + half * 8);       // elements 0..7
    f.q[1] = *(const uint4*)(rowp + 16 + half * 8);  // elements 8..15
    return f.v;
}

// B fragment pre-swizzled in memory: 32B contiguous per lane.
__device__ __forceinline__ v16bf load_frag_b(const unsigned short* p) {
    Frag16 f;
    f.q[0] = *(const uint4*)(p);
    f.q[1] = *(const uint4*)(p + 8);
    return f.v;
}

// ---------------------------------------------------------------------------
// Weight pre-swizzle: f32 (K x N, row major) -> bf16 B-fragment order
// layout: [kt][nt][lane(32)][elem(16)]   k = kt*32 + (lane>>4)*8 + (e&7) + (e>>3)*16
//                                        n = nt*16 + (lane&15)
// rows k >= K (padding) become 0, so padded-K activations contribute nothing.
// ---------------------------------------------------------------------------
__global__ void pack_w(const float* __restrict__ W, int K, int Kpad, int Nn,
                       unsigned short* __restrict__ outp) {
    int i = blockIdx.x * blockDim.x + threadIdx.x;
    int total = Kpad * Nn;
    if (i >= total) return;
    int e    = i & 15;
    int ln   = (i >> 4) & 31;
    int rest = i >> 9;               // kt*NT + nt
    int NT   = Nn >> 4;
    int nt   = rest % NT;
    int kt   = rest / NT;
    int k = kt * 32 + (ln >> 4) * 8 + (e & 7) + (e >> 3) * 16;
    int n = nt * 16 + (ln & 15);
    float v = (k < K) ? W[(size_t)k * Nn + n] : 0.0f;
    outp[i] = f2bf(v);
}

__global__ void zero_out_kernel(float* __restrict__ out, int n) {
    int i = blockIdx.x * blockDim.x + threadIdx.x;
    if (i < n) out[i] = 0.0f;
}

// ---------------------------------------------------------------------------
// Epilogue for one 64x16 column tile: bias + optional relu, packed bf16 cvt.
// C/D element j of lane L: row = 16*mt + 8*(L>>4) + j, col = L&15. (ISA 7.12.2)
// ---------------------------------------------------------------------------
__device__ __forceinline__
void store_tile(const v8f* acc, float bb, unsigned short* __restrict__ dst,
                int strideD, int ncol, int half, bool do_relu) {
#pragma unroll
    for (int mt = 0; mt < 4; ++mt) {
#pragma unroll
        for (int j = 0; j < 8; j += 2) {
            float v0 = acc[mt][j]     + bb;
            float v1 = acc[mt][j + 1] + bb;
            if (do_relu) { v0 = fmaxf(v0, 0.0f); v1 = fmaxf(v1, 0.0f); }
            unsigned int p = cvt_pk_bf16(v0, v1);
            int r = mt * 16 + half * 8 + j;
            dst[(size_t)r       * strideD + ncol] = (unsigned short)(p & 0xFFFFu);
            dst[(size_t)(r + 1) * strideD + ncol] = (unsigned short)(p >> 16);
        }
    }
}

// ---------------------------------------------------------------------------
// One fused GEMM stage: dst(64 x Nout) = act( src(64 x Ks) @ Wswz + bias )
// 4 waves; wave w owns n-tile pairs (w, w+4), (w+8, w+12), ...; all 4 M-tiles
// per wave. Each A fragment feeds TWO wmma (halves LDS traffic), each B
// fragment is fetched once per wave. Requires ntiles % 8 == 0 (8/8/16/32).
// ---------------------------------------------------------------------------
__device__ __forceinline__
void gemm_stage(const unsigned short* __restrict__ src, int ktiles, int strideS,
                const unsigned short* __restrict__ wswz,
                const float* __restrict__ bias,
                unsigned short* __restrict__ dst, int ntiles, int strideD,
                bool do_relu, int lane, int wave) {
    const int col  = lane & 15;
    const int half = lane >> 4;
    for (int nt0 = wave; nt0 < ntiles; nt0 += 8) {
        const int nt1 = nt0 + 4;
        v8f z = {};
        v8f acc0[4] = {z, z, z, z};
        v8f acc1[4] = {z, z, z, z};
        for (int kt = 0; kt < ktiles; ++kt) {
            v16bf b0 = load_frag_b(wswz + ((size_t)(kt * ntiles + nt0) * 32 + lane) * 16);
            v16bf b1 = load_frag_b(wswz + ((size_t)(kt * ntiles + nt1) * 32 + lane) * 16);
#pragma unroll
            for (int mt = 0; mt < 4; ++mt) {
                v16bf a = load_frag_a(src + (mt * 16 + col) * strideS + kt * 32, half);
                acc0[mt] = __builtin_amdgcn_wmma_f32_16x16x32_bf16(
                    false, a, false, b0, (short)0, acc0[mt], false, false);
                acc1[mt] = __builtin_amdgcn_wmma_f32_16x16x32_bf16(
                    false, a, false, b1, (short)0, acc1[mt], false, false);
            }
        }
        store_tile(acc0, bias[nt0 * 16 + col], dst, strideD, nt0 * 16 + col, half, do_relu);
        store_tile(acc1, bias[nt1 * 16 + col], dst, strideD, nt1 * 16 + col, half, do_relu);
    }
}

// Final stage: relu(H @ We2 + be2), fused max over the block's 64 rows,
// atomicMax (f32-as-int, valid: values >= 0) into the leaf feature row.
__device__ __forceinline__
void gemm_max_stage(const unsigned short* __restrict__ src, int ktiles, int strideS,
                    const unsigned short* __restrict__ wswz,
                    const float* __restrict__ bias,
                    float* __restrict__ leafout, int ntiles, int lane, int wave) {
    const int col  = lane & 15;
    const int half = lane >> 4;
    for (int nt0 = wave; nt0 < ntiles; nt0 += 8) {
        const int nt1 = nt0 + 4;
        v8f z = {};
        v8f acc0[4] = {z, z, z, z};
        v8f acc1[4] = {z, z, z, z};
        for (int kt = 0; kt < ktiles; ++kt) {
            v16bf b0 = load_frag_b(wswz + ((size_t)(kt * ntiles + nt0) * 32 + lane) * 16);
            v16bf b1 = load_frag_b(wswz + ((size_t)(kt * ntiles + nt1) * 32 + lane) * 16);
#pragma unroll
            for (int mt = 0; mt < 4; ++mt) {
                v16bf a = load_frag_a(src + (mt * 16 + col) * strideS + kt * 32, half);
                acc0[mt] = __builtin_amdgcn_wmma_f32_16x16x32_bf16(
                    false, a, false, b0, (short)0, acc0[mt], false, false);
                acc1[mt] = __builtin_amdgcn_wmma_f32_16x16x32_bf16(
                    false, a, false, b1, (short)0, acc1[mt], false, false);
            }
        }
        float bb0 = bias[nt0 * 16 + col];
        float bb1 = bias[nt1 * 16 + col];
        float m0 = 0.0f, m1 = 0.0f;     // relu floor folded into the max
#pragma unroll
        for (int mt = 0; mt < 4; ++mt) {
#pragma unroll
            for (int j = 0; j < 8; ++j) {
                m0 = fmaxf(m0, acc0[mt][j] + bb0);
                m1 = fmaxf(m1, acc1[mt][j] + bb1);
            }
        }
        m0 = fmaxf(m0, __shfl_xor(m0, 16, 32));   // combine lane halves (rows 0-7 / 8-15)
        m1 = fmaxf(m1, __shfl_xor(m1, 16, 32));
        if (lane < 16) {
            atomicMax((int*)(leafout + nt0 * 16 + col), __float_as_int(m0));
            atomicMax((int*)(leafout + nt1 * 16 + col), __float_as_int(m1));
        }
    }
}

// ---------------------------------------------------------------------------
// Fused per-point pipeline: gather -> proj MLP -> encoder MLP -> leaf max.
// 2048 blocks x 128 threads; 64 points per block; all intermediates in LDS.
// ---------------------------------------------------------------------------
#define PP_ROWS 64

__global__ __launch_bounds__(128)
void point_pipeline(const float* __restrict__ coords,
                    const float* __restrict__ feats,
                    const int*   __restrict__ leaf_indices,
                    const int*   __restrict__ leaf_center_idx,
                    const unsigned short* __restrict__ wp1s,
                    const unsigned short* __restrict__ wp2s,
                    const unsigned short* __restrict__ we1s,
                    const unsigned short* __restrict__ we2s,
                    const float* __restrict__ bp1,
                    const float* __restrict__ bp2,
                    const float* __restrict__ be1,
                    const float* __restrict__ be2,
                    float* __restrict__ out) {
    extern __shared__ unsigned short smem[];
    unsigned short* smF = smem;                      // 64 x 32  raw feats (bf16)
    unsigned short* smX = smem + PP_ROWS * 32;       // 64 x 160 [rel(3) | proj(128) | pad(29)]
    unsigned short* smB = smX + PP_ROWS * 160;       // 64 x 256 Y1 then H

    const int bid  = blockIdx.x;
    const int leaf = bid >> 5;                       // 32 blocks of 64 rows per 2048-pt leaf
    const int t    = threadIdx.x;
    const int lane = t & 31;
    const int wave = t >> 5;

    // ---- gather phase: 2 threads per row, packed bf16 conversion + b32 LDS stores ----
    {
        int row  = t >> 1;
        int part = t & 1;
        int gr   = bid * PP_ROWS + row;              // flat (leaf, k) row
        int idx  = leaf_indices[gr];
        const float* fr = feats + (size_t)idx * 32 + part * 16;
        unsigned int* frow = (unsigned int*)(smF + row * 32 + part * 16);
#pragma unroll
        for (int i = 0; i < 16; i += 4) {
            float4 v = *(const float4*)(fr + i);
            frow[(i >> 1) + 0] = cvt_pk_bf16(v.x, v.y);
            frow[(i >> 1) + 1] = cvt_pk_bf16(v.z, v.w);
        }
        if (part == 0) {
            int c = leaf_center_idx[leaf];
            float rx = coords[(size_t)idx * 3 + 0] - coords[(size_t)c * 3 + 0];
            float ry = coords[(size_t)idx * 3 + 1] - coords[(size_t)c * 3 + 1];
            float rz = coords[(size_t)idx * 3 + 2] - coords[(size_t)c * 3 + 2];
            unsigned int pxy = cvt_pk_bf16(rx, ry);
            *(unsigned int*)(smX + row * 160) = pxy;       // cols 0,1 (4B aligned: 160*2B rows)
            smX[row * 160 + 2] = (unsigned short)(cvt_pk_bf16(rz, 0.0f) & 0xFFFFu);
        } else {
            for (int c = 131; c < 160; ++c) smX[row * 160 + c] = 0;  // kill NaN garbage
        }
    }
    __syncthreads();

    // Y1 = relu(X0 @ Wp1 + bp1)           (64x32)(32x128)   -> smB (stride 128)
    gemm_stage(smF, 1, 32, wp1s, bp1, smB, 8, 128, true, lane, wave);
    __syncthreads();
    // proj = Y1 @ Wp2 + bp2  (no relu)    (64x128)(128x128) -> smX cols 3..130
    gemm_stage(smB, 4, 128, wp2s, bp2, smX + 3, 8, 160, false, lane, wave);
    __syncthreads();
    // H = relu(X @ We1 + be1)             (64x160)(160x256) -> smB (stride 256)
    gemm_stage(smX, 5, 160, we1s, be1, smB, 16, 256, true, lane, wave);
    __syncthreads();
    // PF = relu(H @ We2 + be2); max over 64 rows; atomicMax into leaf row
    gemm_max_stage(smB, 8, 256, we2s, be2, out + (size_t)(9 + leaf) * 512, 32, lane, wave);
}

// ---------------------------------------------------------------------------
// Tiny aggregator (0.07% of FLOPs): per parent, g_c = relu([cf_c, r_c] @ Wa1
// + ba1), out = max_c g_c @ Wa2 + ba2.  256 threads, 2 output cols each.
// ---------------------------------------------------------------------------
__global__ __launch_bounds__(256)
void agg_kernel(const float* __restrict__ feats_base, int feats_block_stride,
                const float* __restrict__ coords,
                const int*   __restrict__ cidx_base, int cidx_stride,
                const int*   __restrict__ pidx,
                const float* __restrict__ Wa1, const float* __restrict__ ba1,
                const float* __restrict__ Wa2, const float* __restrict__ ba2,
                float* __restrict__ out_base, int out_stride) {
    __shared__ float z[520];
    __shared__ float gmax[512];
    const int t = threadIdx.x;
    const int p = blockIdx.x;
    const int* cidx = cidx_base + p * cidx_stride;
    const int pi = pidx[p];
    const float pxyz[3] = { coords[(size_t)pi * 3 + 0],
                            coords[(size_t)pi * 3 + 1],
                            coords[(size_t)pi * 3 + 2] };
    gmax[t] = 0.0f;
    gmax[t + 256] = 0.0f;
    for (int c = 0; c < 8; ++c) {
        const float* cf = feats_base + (size_t)p * feats_block_stride + (size_t)c * 512;
        z[t]       = cf[t];
        z[t + 256] = cf[t + 256];
        if (t < 3) {
            int ci = cidx[c];
            z[512 + t] = coords[(size_t)ci * 3 + t] - pxyz[t];
        }
        __syncthreads();
        float a0 = ba1[t], a1 = ba1[t + 256];
        for (int k = 0; k < 515; ++k) {
            float zk = z[k];
            a0 += zk * Wa1[(size_t)k * 512 + t];
            a1 += zk * Wa1[(size_t)k * 512 + t + 256];
        }
        gmax[t]       = fmaxf(gmax[t],       fmaxf(a0, 0.0f));
        gmax[t + 256] = fmaxf(gmax[t + 256], fmaxf(a1, 0.0f));
        __syncthreads();
    }
    float a0 = ba2[t], a1 = ba2[t + 256];
    for (int k = 0; k < 512; ++k) {
        float g = gmax[k];
        a0 += g * Wa2[(size_t)k * 512 + t];
        a1 += g * Wa2[(size_t)k * 512 + t + 256];
    }
    out_base[(size_t)p * out_stride + t]       = a0;
    out_base[(size_t)p * out_stride + t + 256] = a1;
}

// ---------------------------------------------------------------------------
extern "C" void kernel_launch(void* const* d_in, const int* in_sizes, int n_in,
                              void* d_out, int out_size, void* d_ws, size_t ws_size,
                              hipStream_t stream) {
    (void)in_sizes; (void)n_in; (void)out_size; (void)ws_size;
    const float* coords          = (const float*)d_in[0];
    const float* feats           = (const float*)d_in[1];
    const int*   leaf_indices    = (const int*)d_in[2];
    const int*   leaf_center_idx = (const int*)d_in[3];
    const int*   l1_center_idx   = (const int*)d_in[4];
    const int*   root_center_idx = (const int*)d_in[5];
    const float* Wp1 = (const float*)d_in[6];  const float* bp1 = (const float*)d_in[7];
    const float* Wp2 = (const float*)d_in[8];  const float* bp2 = (const float*)d_in[9];
    const float* We1 = (const float*)d_in[10]; const float* be1 = (const float*)d_in[11];
    const float* We2 = (const float*)d_in[12]; const float* be2 = (const float*)d_in[13];
    const float* Wa1 = (const float*)d_in[14]; const float* ba1 = (const float*)d_in[15];
    const float* Wa2 = (const float*)d_in[16]; const float* ba2 = (const float*)d_in[17];
    float* out = (float*)d_out;

    // workspace: packed bf16 weights (377 KB total)
    unsigned short* wp1s = (unsigned short*)d_ws;        // 32*128
    unsigned short* wp2s = wp1s + 32 * 128;              // 128*128
    unsigned short* we1s = wp2s + 128 * 128;             // 160*256 (K padded 131->160)
    unsigned short* we2s = we1s + 160 * 256;             // 256*512

    const int OUT_ELEMS = 73 * 512;
    zero_out_kernel<<<(OUT_ELEMS + 255) / 256, 256, 0, stream>>>(out, OUT_ELEMS);

    pack_w<<<(32 * 128 + 255) / 256, 256, 0, stream>>>(Wp1, 32, 32, 128, wp1s);
    pack_w<<<(128 * 128 + 255) / 256, 256, 0, stream>>>(Wp2, 128, 128, 128, wp2s);
    pack_w<<<(160 * 256 + 255) / 256, 256, 0, stream>>>(We1, 131, 160, 256, we1s);
    pack_w<<<(256 * 512 + 255) / 256, 256, 0, stream>>>(We2, 256, 256, 512, we2s);

    // 131072 points / 64 rows per block
    size_t lds_bytes = (size_t)(PP_ROWS * 32 + PP_ROWS * 160 + PP_ROWS * 256) * sizeof(unsigned short);
    point_pipeline<<<2048, 128, lds_bytes, stream>>>(
        coords, feats, leaf_indices, leaf_center_idx,
        wp1s, wp2s, we1s, we2s, bp1, bp2, be1, be2, out);

    // level 1: 8 parents, each aggregates 8 leaf rows (out rows 9..72)
    agg_kernel<<<8, 256, 0, stream>>>(out + 9 * 512, 8 * 512, coords,
                                      leaf_center_idx, 8, l1_center_idx,
                                      Wa1, ba1, Wa2, ba2, out + 512, 512);
    // root: aggregates the 8 level-1 rows (out rows 1..8) -> out row 0
    agg_kernel<<<1, 256, 0, stream>>>(out + 512, 0, coords,
                                      l1_center_idx, 0, root_center_idx,
                                      Wa1, ba1, Wa2, ba2, out, 0);
}
